// MIL_Cluster_FC_ill_61795989455391
// MI455X (gfx1250) — compile-verified, matrix operand order checked by reading, not run on hardware
//
#include <hip/hip_runtime.h>

#define NCLU 10
#define DIN  1024
#define DH   512
#define DATT 256
#define NCLS 4

typedef __attribute__((ext_vector_type(16))) __bf16 v16bf;
typedef __attribute__((ext_vector_type(8)))  float  v8f;

union FragBF {
    unsigned short s[16];
    uint4          q[2];
    v16bf          v;
};

static __device__ __forceinline__ unsigned short f2bf(float f) {
    // round-to-nearest-even f32 -> bf16
    unsigned u = __float_as_uint(f);
    unsigned r = u + 0x7FFFu + ((u >> 16) & 1u);
    return (unsigned short)(r >> 16);
}

// ---------------------------------------------------------------------------
// K0: zero the pooled accumulator
// ---------------------------------------------------------------------------
__global__ void mil_zero_kernel(float* __restrict__ pooled) {
    int t = blockIdx.x * blockDim.x + threadIdx.x;
    if (t < NCLU * DH) pooled[t] = 0.0f;
}

// ---------------------------------------------------------------------------
// K1: histogram + exclusive scan + deterministic per-cluster compaction
// ---------------------------------------------------------------------------
__global__ __launch_bounds__(512) void mil_compact_kernel(
    const int* __restrict__ cid, int* __restrict__ counts,
    int* __restrict__ offs, int* __restrict__ idxl, int N)
{
    __shared__ int lc[NCLU];
    __shared__ int lo[NCLU];
    const int tid = threadIdx.x;
    if (tid < NCLU) lc[tid] = 0;
    __syncthreads();
    for (int n = tid; n < N; n += 512) {
        int cc = cid[n];
        if (cc >= 0 && cc < NCLU) atomicAdd(&lc[cc], 1);
    }
    __syncthreads();
    if (tid == 0) {
        int accum = 0;
        for (int i = 0; i < NCLU; ++i) {
            counts[i] = lc[i];
            lo[i] = accum;
            offs[i] = accum;
            accum += lc[i];
        }
    }
    __syncthreads();
    const int w = tid >> 5, lane = tid & 31;
    if (w < NCLU) {
        int base = lo[w];
        for (int n0 = 0; n0 < N; n0 += 32) {
            int n = n0 + lane;
            bool p = (n < N) && (cid[n] == w);
            unsigned mask = (unsigned)__ballot((int)p);
            if (p) {
                int rank = __popc(mask & ((1u << lane) - 1u));
                idxl[base + rank] = n;
            }
            base += __popc(mask);
        }
    }
}

// ---------------------------------------------------------------------------
// K1b: convert f32 weights -> bf16 pre-swizzled into B-fragment order:
//   dst[((c*NT + n_tile)*KK + kk)*512 + lane*16 + h]
//   n  = n_tile*16 + (lane&15) ; kb = kk*32 + (lane>>4)*8
//   k  = (h<8) ? kb+h : kb+8+h
// ---------------------------------------------------------------------------
__global__ __launch_bounds__(256) void mil_swz_kernel(
    const float* __restrict__ W, unsigned short* __restrict__ dst,
    int C, int K, int N)
{
    const int KK = K / 32, NT = N / 16;
    const long long total = (long long)C * NT * KK * 32;
    long long t = (long long)blockIdx.x * blockDim.x + threadIdx.x;
    if (t >= total) return;
    int lane = (int)(t & 31);
    long long tile = t >> 5;
    int kk = (int)(tile % KK);
    long long ct = tile / KK;
    int n_tile = (int)(ct % NT);
    int c = (int)(ct / NT);

    int n  = n_tile * 16 + (lane & 15);
    int kb = kk * 32 + (lane >> 4) * 8;
    const float* src = W + ((size_t)c * K) * N + n;

    FragBF f;
    #pragma unroll
    for (int h = 0; h < 8; ++h)  f.s[h]     = f2bf(src[(size_t)(kb + h) * N]);
    #pragma unroll
    for (int h = 0; h < 8; ++h)  f.s[8 + h] = f2bf(src[(size_t)(kb + 16 + h) * N]);

    uint4* out = (uint4*)(dst + tile * 512 + (size_t)lane * 16);
    out[0] = f.q[0];
    out[1] = f.q[1];
}

// ---------------------------------------------------------------------------
// K2 (fast path): M=32 tile, pre-swizzled bf16 weights, fragment-order LDS.
// grid = (ceil(N/32), NCLU), block = 256 (8 waves).
// Wave owns a 32(M) x 64(N) slab: acc[2][4], B frag reused across 2 m-tiles.
// A-frag loads are lane-consecutive 32B (bank-conflict free).
// H1 (32KB, frag order) aliases the X tile region (64KB) after a barrier.
// ---------------------------------------------------------------------------
__global__ __launch_bounds__(256) void mil_expert_swz_kernel(
    const float* __restrict__ xp,
    const unsigned short* __restrict__ W1s, const float* __restrict__ b1,
    const unsigned short* __restrict__ W2s, const float* __restrict__ b2,
    const int* __restrict__ counts, const int* __restrict__ offs,
    const int* __restrict__ idxl, float* __restrict__ pooled)
{
    const int c   = blockIdx.y;
    const int cnt = counts[c];
    const int m0  = blockIdx.x * 32;
    if (m0 >= cnt) return;
    const int off = offs[c];

    const int KK1 = DIN / 32;   // 32 k-tiles, layer 1
    const int KK2 = DH  / 32;   // 16 k-tiles, layer 2
    const int NT  = DH  / 16;   // 32 n-tiles

    __shared__ __align__(16) unsigned short Xf[32 * DIN]; // 64 KB, frag order
    unsigned short* Hf = Xf;                               // aliased after sync

    const int tid = threadIdx.x;

    // ---- stage X tile (32 rows) f32 -> bf16 directly in A-fragment order ---
    for (int j = tid; j < 2 * KK1 * 32; j += 256) {
        int lane_j = j & 31;
        int tile   = j >> 5;              // mt*KK1 + kk
        int mt     = tile / KK1;
        int kkv    = tile % KK1;
        int m  = mt * 16 + (lane_j & 15);
        int kb = kkv * 32 + (lane_j >> 4) * 8;
        int gm = m0 + m;
        int ri = (gm < cnt) ? idxl[off + gm] : -1;
        FragBF f;
        if (ri >= 0) {
            const float* src = xp + (size_t)ri * DIN;
            float4 v0 = *(const float4*)(src + kb);
            float4 v1 = *(const float4*)(src + kb + 4);
            float4 v2 = *(const float4*)(src + kb + 16);
            float4 v3 = *(const float4*)(src + kb + 20);
            f.s[0]  = f2bf(v0.x); f.s[1]  = f2bf(v0.y);
            f.s[2]  = f2bf(v0.z); f.s[3]  = f2bf(v0.w);
            f.s[4]  = f2bf(v1.x); f.s[5]  = f2bf(v1.y);
            f.s[6]  = f2bf(v1.z); f.s[7]  = f2bf(v1.w);
            f.s[8]  = f2bf(v2.x); f.s[9]  = f2bf(v2.y);
            f.s[10] = f2bf(v2.z); f.s[11] = f2bf(v2.w);
            f.s[12] = f2bf(v3.x); f.s[13] = f2bf(v3.y);
            f.s[14] = f2bf(v3.z); f.s[15] = f2bf(v3.w);
        } else {
            f.q[0] = make_uint4(0, 0, 0, 0);
            f.q[1] = make_uint4(0, 0, 0, 0);
        }
        uint4* dst = (uint4*)&Xf[(size_t)j * 16];
        dst[0] = f.q[0];
        dst[1] = f.q[1];
    }
    __syncthreads();

    const int lane  = tid & 31;
    const int wave  = tid >> 5;
    const int lhalf = lane >> 4;
    const int lcol  = lane & 15;
    const int nb0   = wave * 64;
    const int mb    = lhalf * 8;

    const float* b1c = b1 + c * DH;
    const float* b2c = b2 + c * DH;

    v8f acc[2][4];

    // =================== Layer 1: H1 = relu(X @ W1 + b1) ===================
    {
        const unsigned short* Wbase =
            W1s + (((size_t)c * NT) * KK1) * 512 + (size_t)lane * 16;

        #pragma unroll
        for (int nt = 0; nt < 4; ++nt) {
            float bv = b1c[nb0 + nt * 16 + lcol];
            #pragma unroll
            for (int r = 0; r < 8; ++r) { acc[0][nt][r] = bv; acc[1][nt][r] = bv; }
        }
        for (int kk = 0; kk < KK1; ++kk) {
            FragBF a0, a1;
            const uint4* pa0 = (const uint4*)&Xf[((size_t)kk * 32 + lane) * 16];
            const uint4* pa1 = (const uint4*)&Xf[(((size_t)KK1 + kk) * 32 + lane) * 16];
            a0.q[0] = pa0[0]; a0.q[1] = pa0[1];
            a1.q[0] = pa1[0]; a1.q[1] = pa1[1];
            #pragma unroll
            for (int nt = 0; nt < 4; ++nt) {
                int ntile = wave * 4 + nt;
                const uint4* wp =
                    (const uint4*)(Wbase + ((size_t)ntile * KK1 + kk) * 512);
                FragBF b;
                b.q[0] = wp[0];
                b.q[1] = wp[1];
                acc[0][nt] = __builtin_amdgcn_wmma_f32_16x16x32_bf16(
                    false, a0.v, false, b.v, (short)0, acc[0][nt], false, false);
                acc[1][nt] = __builtin_amdgcn_wmma_f32_16x16x32_bf16(
                    false, a1.v, false, b.v, (short)0, acc[1][nt], false, false);
            }
        }
    }
    __syncthreads();   // all waves done reading Xf before aliasing as Hf

    // ---- relu + store H1 into LDS in A-fragment order for layer 2 ----
    {
        #pragma unroll
        for (int mt = 0; mt < 2; ++mt) {
            #pragma unroll
            for (int nt = 0; nt < 4; ++nt) {
                int col = nb0 + nt * 16 + lcol;
                int kk2 = col >> 5;
                int r32 = col & 31;
                int s   = (r32 >> 3) & 1;
                int h   = (r32 & 7) | ((r32 >> 4) << 3);
                #pragma unroll
                for (int r = 0; r < 8; ++r) {
                    int m = mt * 16 + mb + r;
                    int lane2 = (m & 15) | (s << 4);
                    size_t di = (((size_t)mt * KK2 + kk2) * 32 + lane2) * 16 + h;
                    Hf[di] = f2bf(fmaxf(acc[mt][nt][r], 0.0f));
                }
            }
        }
    }
    __syncthreads();

    // =================== Layer 2: H2 = relu(H1 @ W2 + b2) ==================
    {
        const unsigned short* Wbase =
            W2s + (((size_t)c * NT) * KK2) * 512 + (size_t)lane * 16;

        #pragma unroll
        for (int nt = 0; nt < 4; ++nt) {
            float bv = b2c[nb0 + nt * 16 + lcol];
            #pragma unroll
            for (int r = 0; r < 8; ++r) { acc[0][nt][r] = bv; acc[1][nt][r] = bv; }
        }
        for (int kk = 0; kk < KK2; ++kk) {
            FragBF a0, a1;
            const uint4* pa0 = (const uint4*)&Hf[((size_t)kk * 32 + lane) * 16];
            const uint4* pa1 = (const uint4*)&Hf[(((size_t)KK2 + kk) * 32 + lane) * 16];
            a0.q[0] = pa0[0]; a0.q[1] = pa0[1];
            a1.q[0] = pa1[0]; a1.q[1] = pa1[1];
            #pragma unroll
            for (int nt = 0; nt < 4; ++nt) {
                int ntile = wave * 4 + nt;
                const uint4* wp =
                    (const uint4*)(Wbase + ((size_t)ntile * KK2 + kk) * 512);
                FragBF b;
                b.q[0] = wp[0];
                b.q[1] = wp[1];
                acc[0][nt] = __builtin_amdgcn_wmma_f32_16x16x32_bf16(
                    false, a0.v, false, b.v, (short)0, acc[0][nt], false, false);
                acc[1][nt] = __builtin_amdgcn_wmma_f32_16x16x32_bf16(
                    false, a1.v, false, b.v, (short)0, acc[1][nt], false, false);
            }
        }
    }

    // ---- relu + masked column-sum over valid rows, accumulate pooled ----
    {
        #pragma unroll
        for (int nt = 0; nt < 4; ++nt) {
            float part = 0.0f;
            #pragma unroll
            for (int mt = 0; mt < 2; ++mt) {
                #pragma unroll
                for (int r = 0; r < 8; ++r) {
                    bool valid = (m0 + mt * 16 + mb + r) < cnt;
                    float v = fmaxf(acc[mt][nt][r], 0.0f);
                    part += valid ? v : 0.0f;
                }
            }
            float tot = part + __shfl_xor(part, 16);
            if (lhalf == 0)
                atomicAdd(&pooled[c * DH + nb0 + nt * 16 + lcol], tot);
        }
    }
}

// ---------------------------------------------------------------------------
// K2 (fallback path): M=16, inline f32->bf16 weight conversion (only if d_ws
// is too small for the swizzled copies).
// ---------------------------------------------------------------------------
__global__ __launch_bounds__(256) void mil_expert_fallback_kernel(
    const float* __restrict__ xp,
    const float* __restrict__ W1, const float* __restrict__ b1,
    const float* __restrict__ W2, const float* __restrict__ b2,
    const int* __restrict__ counts, const int* __restrict__ offs,
    const int* __restrict__ idxl, float* __restrict__ pooled)
{
    const int c   = blockIdx.y;
    const int cnt = counts[c];
    const int m0  = blockIdx.x * 16;
    if (m0 >= cnt) return;
    const int off = offs[c];

    __shared__ unsigned short Xs[16 * DIN];
    __shared__ unsigned short Hs[16 * DH];
    __shared__ int ridx[16];

    const int tid = threadIdx.x;
    if (tid < 16) {
        int m = m0 + tid;
        ridx[tid] = (m < cnt) ? idxl[off + m] : -1;
    }
    __syncthreads();
    for (int t = tid; t < 16 * (DIN / 4); t += 256) {
        int r  = t / (DIN / 4);
        int c4 = (t % (DIN / 4)) * 4;
        int ri = ridx[r];
        float4 v;
        if (ri >= 0) v = *(const float4*)(xp + (size_t)ri * DIN + c4);
        else         v = make_float4(0.f, 0.f, 0.f, 0.f);
        unsigned p0 = (unsigned)f2bf(v.x) | ((unsigned)f2bf(v.y) << 16);
        unsigned p1 = (unsigned)f2bf(v.z) | ((unsigned)f2bf(v.w) << 16);
        *(uint2*)&Xs[r * DIN + c4] = make_uint2(p0, p1);
    }
    __syncthreads();

    const int wave  = tid >> 5;
    const int lane  = tid & 31;
    const int lhalf = lane >> 4;
    const int lcol  = lane & 15;
    const int nb0   = wave * 64;

    const float* W1c = W1 + (size_t)c * DIN * DH;
    const float* b1c = b1 + c * DH;
    const float* W2c = W2 + (size_t)c * DH * DH;
    const float* b2c = b2 + c * DH;

    v8f acc[4];

    #pragma unroll
    for (int nt = 0; nt < 4; ++nt) {
        float bv = b1c[nb0 + nt * 16 + lcol];
        #pragma unroll
        for (int r = 0; r < 8; ++r) acc[nt][r] = bv;
    }
    for (int kk = 0; kk < DIN / 32; ++kk) {
        int kb = kk * 32 + lhalf * 8;
        FragBF a;
        a.q[0] = *(const uint4*)&Xs[lcol * DIN + kb];
        a.q[1] = *(const uint4*)&Xs[lcol * DIN + kb + 16];
        #pragma unroll
        for (int nt = 0; nt < 4; ++nt) {
            int n = nb0 + nt * 16 + lcol;
            const float* wp = W1c + (size_t)kb * DH + n;
            FragBF b;
            #pragma unroll
            for (int i = 0; i < 8; ++i) b.s[i]     = f2bf(wp[i * DH]);
            #pragma unroll
            for (int i = 0; i < 8; ++i) b.s[8 + i] = f2bf(wp[(16 + i) * DH]);
            acc[nt] = __builtin_amdgcn_wmma_f32_16x16x32_bf16(
                false, a.v, false, b.v, (short)0, acc[nt], false, false);
        }
    }
    {
        int mb = lhalf * 8;
        #pragma unroll
        for (int nt = 0; nt < 4; ++nt) {
            int col = nb0 + nt * 16 + lcol;
            #pragma unroll
            for (int r = 0; r < 8; ++r) {
                float v = fmaxf(acc[nt][r], 0.0f);
                Hs[(mb + r) * DH + col] = f2bf(v);
            }
        }
    }
    __syncthreads();

    #pragma unroll
    for (int nt = 0; nt < 4; ++nt) {
        float bv = b2c[nb0 + nt * 16 + lcol];
        #pragma unroll
        for (int r = 0; r < 8; ++r) acc[nt][r] = bv;
    }
    for (int kk = 0; kk < DH / 32; ++kk) {
        int kb = kk * 32 + lhalf * 8;
        FragBF a;
        a.q[0] = *(const uint4*)&Hs[lcol * DH + kb];
        a.q[1] = *(const uint4*)&Hs[lcol * DH + kb + 16];
        #pragma unroll
        for (int nt = 0; nt < 4; ++nt) {
            int n = nb0 + nt * 16 + lcol;
            const float* wp = W2c + (size_t)kb * DH + n;
            FragBF b;
            #pragma unroll
            for (int i = 0; i < 8; ++i) b.s[i]     = f2bf(wp[i * DH]);
            #pragma unroll
            for (int i = 0; i < 8; ++i) b.s[8 + i] = f2bf(wp[(16 + i) * DH]);
            acc[nt] = __builtin_amdgcn_wmma_f32_16x16x32_bf16(
                false, a.v, false, b.v, (short)0, acc[nt], false, false);
        }
    }
    {
        int mb = lhalf * 8;
        #pragma unroll
        for (int nt = 0; nt < 4; ++nt) {
            float part = 0.0f;
            #pragma unroll
            for (int r = 0; r < 8; ++r) {
                bool valid = (m0 + mb + r) < cnt;
                float v = fmaxf(acc[nt][r], 0.0f);
                part += valid ? v : 0.0f;
            }
            float tot = part + __shfl_xor(part, 16);
            if (lhalf == 0)
                atomicAdd(&pooled[c * DH + nb0 + nt * 16 + lcol], tot);
        }
    }
}

// ---------------------------------------------------------------------------
// K3: attention head + rho + classifier + hazards/survival/argmax
// ---------------------------------------------------------------------------
__global__ __launch_bounds__(256) void mil_head_kernel(
    const float* __restrict__ pooled, const int* __restrict__ counts,
    const float* __restrict__ Wfc, const float* __restrict__ bfc,
    const float* __restrict__ Wa,  const float* __restrict__ ba,
    const float* __restrict__ Wb,  const float* __restrict__ bb,
    const float* __restrict__ Wc,  const float* __restrict__ bc,
    const float* __restrict__ Wrho, const float* __restrict__ brho,
    const float* __restrict__ Wcls, const float* __restrict__ bcls,
    float* __restrict__ out)
{
    __shared__ float hc[NCLU * DH];
    __shared__ float hh[NCLU * DH];
    __shared__ float tt[NCLU * DATT];
    __shared__ float Aw[NCLU];
    __shared__ float hp[DH];
    __shared__ float hr[DATT];
    __shared__ float lg[NCLS];
    const int tid = threadIdx.x;

    for (int i = tid; i < NCLU * DH; i += 256) {
        int ci = i / DH;
        float cntf = fmaxf((float)counts[ci], 1.0f);
        hc[i] = pooled[i] / cntf;
    }
    __syncthreads();
    for (int i = tid; i < NCLU * DH; i += 256) {
        int ci = i / DH, j = i % DH;
        float s = bfc[j];
        const float* row = &hc[ci * DH];
        for (int k = 0; k < DH; ++k) s += row[k] * Wfc[k * DH + j];
        hh[i] = fmaxf(s, 0.0f);
    }
    __syncthreads();
    for (int i = tid; i < NCLU * DATT; i += 256) {
        int ci = i / DATT, j = i % DATT;
        float sa = ba[j], sb = bb[j];
        const float* row = &hh[ci * DH];
        for (int k = 0; k < DH; ++k) {
            float hv = row[k];
            sa += hv * Wa[k * DATT + j];
            sb += hv * Wb[k * DATT + j];
        }
        tt[i] = tanhf(sa) * (1.0f / (1.0f + expf(-sb)));
    }
    __syncthreads();
    if (tid < NCLU) {
        float s = bc[0];
        const float* row = &tt[tid * DATT];
        for (int k = 0; k < DATT; ++k) s += row[k] * Wc[k];
        Aw[tid] = s;
    }
    __syncthreads();
    if (tid == 0) {
        float mx = Aw[0];
        for (int i = 1; i < NCLU; ++i) mx = fmaxf(mx, Aw[i]);
        float sum = 0.0f;
        for (int i = 0; i < NCLU; ++i) { Aw[i] = expf(Aw[i] - mx); sum += Aw[i]; }
        float inv = 1.0f / sum;
        for (int i = 0; i < NCLU; ++i) Aw[i] *= inv;
    }
    __syncthreads();
    for (int j = tid; j < DH; j += 256) {
        float s = 0.0f;
        for (int ci = 0; ci < NCLU; ++ci) s += Aw[ci] * hh[ci * DH + j];
        hp[j] = s;
    }
    __syncthreads();
    for (int j = tid; j < DATT; j += 256) {
        float s = brho[j];
        for (int k = 0; k < DH; ++k) s += hp[k] * Wrho[k * DATT + j];
        hr[j] = fmaxf(s, 0.0f);
    }
    __syncthreads();
    if (tid < NCLS) {
        float s = bcls[tid];
        for (int k = 0; k < DATT; ++k) s += hr[k] * Wcls[k * NCLS + tid];
        lg[tid] = s;
    }
    __syncthreads();
    if (tid == 0) {
        float Sprod = 1.0f;
        int arg = 0; float best = lg[0];
        for (int k = 0; k < NCLS; ++k) {
            float hz = 1.0f / (1.0f + expf(-lg[k]));
            out[k] = hz;
            Sprod *= (1.0f - hz);
            out[NCLS + k] = Sprod;
            if (lg[k] > best) { best = lg[k]; arg = k; }
        }
        out[2 * NCLS] = (float)arg;
    }
}

// ---------------------------------------------------------------------------
static inline size_t align_up(size_t v, size_t a) { return (v + a - 1) & ~(a - 1); }

extern "C" void kernel_launch(void* const* d_in, const int* in_sizes, int n_in,
                              void* d_out, int out_size, void* d_ws, size_t ws_size,
                              hipStream_t stream) {
    const float* xp   = (const float*)d_in[0];
    const int*   cid  = (const int*)d_in[1];
    const float* W1   = (const float*)d_in[2];
    const float* b1   = (const float*)d_in[3];
    const float* W2   = (const float*)d_in[4];
    const float* b2   = (const float*)d_in[5];
    const float* Wfc  = (const float*)d_in[6];
    const float* bfc  = (const float*)d_in[7];
    const float* Wa   = (const float*)d_in[8];
    const float* ba   = (const float*)d_in[9];
    const float* Wb   = (const float*)d_in[10];
    const float* bb   = (const float*)d_in[11];
    const float* Wc   = (const float*)d_in[12];
    const float* bc   = (const float*)d_in[13];
    const float* Wrho = (const float*)d_in[14];
    const float* brho = (const float*)d_in[15];
    const float* Wcls = (const float*)d_in[16];
    const float* bcls = (const float*)d_in[17];
    const int N = in_sizes[1];

    // workspace layout
    uintptr_t base = (uintptr_t)d_ws;
    size_t o = 0;
    int* counts = (int*)(base + o);            o += 16 * sizeof(int);
    int* offs   = (int*)(base + o);            o += 16 * sizeof(int);
    int* idxl   = (int*)(base + o);            o += (size_t)N * sizeof(int);
    float* pooled = (float*)(base + o);        o += NCLU * DH * sizeof(float);
    o = align_up(o, 32);
    unsigned short* W1s = (unsigned short*)(base + o);
    size_t w1_elems = (size_t)NCLU * DIN * DH; o += w1_elems * sizeof(unsigned short);
    unsigned short* W2s = (unsigned short*)(base + o);
    size_t w2_elems = (size_t)NCLU * DH * DH;  o += w2_elems * sizeof(unsigned short);
    const bool use_swz = (ws_size >= o);

    mil_zero_kernel<<<(NCLU * DH + 255) / 256, 256, 0, stream>>>(pooled);
    mil_compact_kernel<<<1, 512, 0, stream>>>(cid, counts, offs, idxl, N);

    if (use_swz) {
        long long t1 = (long long)NCLU * (DH / 16) * (DIN / 32) * 32;
        long long t2 = (long long)NCLU * (DH / 16) * (DH / 32) * 32;
        mil_swz_kernel<<<(unsigned)((t1 + 255) / 256), 256, 0, stream>>>(
            W1, W1s, NCLU, DIN, DH);
        mil_swz_kernel<<<(unsigned)((t2 + 255) / 256), 256, 0, stream>>>(
            W2, W2s, NCLU, DH, DH);
        dim3 grid((N + 31) / 32, NCLU);
        mil_expert_swz_kernel<<<grid, 256, 0, stream>>>(
            xp, W1s, b1, W2s, b2, counts, offs, idxl, pooled);
    } else {
        dim3 grid((N + 15) / 16, NCLU);
        mil_expert_fallback_kernel<<<grid, 256, 0, stream>>>(
            xp, W1, b1, W2, b2, counts, offs, idxl, pooled);
    }

    mil_head_kernel<<<1, 256, 0, stream>>>(pooled, counts, Wfc, bfc, Wa, ba,
                                           Wb, bb, Wc, bc, Wrho, brho,
                                           Wcls, bcls, (float*)d_out);
}
